// LSTM_24867860644472
// MI455X (gfx1250) — compile-verified, matrix operand order checked by reading, not run on hardware
//
#include <hip/hip_runtime.h>
#include <hip/hip_bf16.h>
#include <math.h>

// LSTM: N=64, T=512, D=1024, H=1024, G=4H=4096
#define LN 64
#define LT 512
#define LD 1024
#define LH 1024
#define LG 4096

typedef __attribute__((ext_vector_type(16))) __bf16 v16bf;
typedef __attribute__((ext_vector_type(8)))  float  v8f;

union FragU { uint4 u[2]; v16bf v; };

__device__ __forceinline__ unsigned short f2bf(float f) {
    unsigned int u = __float_as_uint(f);
    unsigned int r = u + 0x7FFFu + ((u >> 16) & 1u);   // round-to-nearest-even
    return (unsigned short)(r >> 16);
}

__device__ __forceinline__ float sigmoidf_(float x) {
    return 1.0f / (1.0f + __expf(-x));
}

__device__ __forceinline__ FragU load_frag(const unsigned short* base, int lk) {
    FragU f;
    f.u[0] = *(const uint4*)(base + lk);
    f.u[1] = *(const uint4*)(base + lk + 16);
    return f;
}

__device__ __forceinline__ v8f wmma_bf16(const FragU& a, const FragU& b, v8f c) {
    return __builtin_amdgcn_wmma_f32_16x16x32_bf16(
        false, a.v, false, b.v, (short)0, c, false, false);
}

// ---------------------------------------------------------------------------
// float -> bf16 flat convert
// ---------------------------------------------------------------------------
__global__ void cvt_f32_bf16_kernel(const float* __restrict__ src,
                                    unsigned short* __restrict__ dst, int n) {
    int i = blockIdx.x * blockDim.x + threadIdx.x;
    if (i < n) dst[i] = f2bf(src[i]);
}

// ---------------------------------------------------------------------------
// (R x C) f32 -> (C x R) bf16 transpose through LDS (32x32 tiles)
// blockDim = (32, 8)
// ---------------------------------------------------------------------------
__global__ void transpose_cvt_kernel(const float* __restrict__ src,
                                     unsigned short* __restrict__ dst,
                                     int R, int C) {
    __shared__ unsigned short tile[32][33];
    int c0 = blockIdx.x * 32;
    int r0 = blockIdx.y * 32;
    for (int i = threadIdx.y; i < 32; i += 8)
        tile[i][threadIdx.x] = f2bf(src[(size_t)(r0 + i) * C + c0 + threadIdx.x]);
    __syncthreads();
    for (int i = threadIdx.y; i < 32; i += 8)
        dst[(size_t)(c0 + i) * R + r0 + threadIdx.x] = tile[threadIdx.x][i];
}

// ---------------------------------------------------------------------------
// xW = x(bf16, M x K) @ WxT(bf16, G x K)^T + bias -> f32 (M x G)
// One wave computes a 16x64 output tile (A-fragment reused across 4 WMMAs).
// K loop fully unrolled: fresh SSA fragments per phase -> no backedge copies,
// scheduler free to overlap each phase's loads with prior WMMAs.
// M = 32768, K = 1024, G = 4096. Grid: 16384 blocks x 256 threads (8 waves).
// ---------------------------------------------------------------------------
__global__ void gemm_xw_kernel(const unsigned short* __restrict__ A,
                               const unsigned short* __restrict__ BT,
                               const float* __restrict__ bias,
                               float* __restrict__ Cout) {
    const int K = LD, G = LG;
    int tid  = threadIdx.x;
    int lane = tid & 31;
    int wid  = blockIdx.x * (blockDim.x >> 5) + (tid >> 5);
    int mt = wid >> 6;                 // 0..2047
    int ng = wid & 63;                 // 0..63
    int m0 = mt * 16;
    int n0 = ng * 64;
    int ksel = (lane >> 4) * 8;        // lanes 16-31 hold the +8 K-halves

    const unsigned short* arow = A + (size_t)(m0 + (lane & 15)) * K;
    const unsigned short* brow[4];
#pragma unroll
    for (int t = 0; t < 4; ++t)
        brow[t] = BT + (size_t)(n0 + t * 16 + (lane & 15)) * K;

    v8f acc[4] = {};
#pragma unroll
    for (int kb = 0; kb < K; kb += 32) {
        if ((kb & 127) == 0)
            __builtin_prefetch(arow + kb + 256, 0, 3);
        int lk = kb + ksel;
        FragU a = load_frag(arow, lk);
        FragU b[4];
#pragma unroll
        for (int t = 0; t < 4; ++t) b[t] = load_frag(brow[t], lk);
#pragma unroll
        for (int t = 0; t < 4; ++t) acc[t] = wmma_bf16(a, b[t], acc[t]);
    }

#pragma unroll
    for (int t = 0; t < 4; ++t) {
        int n = n0 + t * 16 + (lane & 15);
        float bv = bias[n];
#pragma unroll
        for (int r = 0; r < 8; ++r) {
            int m = m0 + r + ((lane >> 4) << 3);
            Cout[(size_t)m * G + n] = acc[t][r] + bv;
        }
    }
}

// ---------------------------------------------------------------------------
// One fused LSTM timestep:
//   a = xw[:, t, :] + h_prev @ Wh ; gates ; c,h update ; store h (f32 + bf16)
// Wave computes 16x16 tiles for all 4 gates at the same (m, hh) tile, so the
// pointwise math happens in-register. K loop fully unrolled.
// Grid: 32 blocks x 256 threads = 256 waves.
// ---------------------------------------------------------------------------
__global__ void lstm_step_kernel(const unsigned short* __restrict__ hbf,  // N x H bf16
                                 const unsigned short* __restrict__ whT,  // G x H bf16
                                 const float* __restrict__ xw,            // N x T x G
                                 float* __restrict__ cbuf,                // N x H
                                 float* __restrict__ out,                 // N x T x H
                                 unsigned short* __restrict__ hout,       // N x H bf16
                                 int t) {
    const int K = LH, H = LH, T = LT;
    int tid  = threadIdx.x;
    int lane = tid & 31;
    int wid  = blockIdx.x * (blockDim.x >> 5) + (tid >> 5);  // 0..255
    int mt  = wid >> 6;    // 0..3
    int hht = wid & 63;    // 0..63
    int m0  = mt * 16;
    int ksel = (lane >> 4) * 8;
    int ncol = hht * 16 + (lane & 15);

    const unsigned short* arow = hbf + (size_t)(m0 + (lane & 15)) * K;
    const unsigned short* brow[4];
#pragma unroll
    for (int g = 0; g < 4; ++g)
        brow[g] = whT + (size_t)(g * H + ncol) * K;

    v8f acc[4] = {};
#pragma unroll
    for (int kb = 0; kb < K; kb += 32) {
        int lk = kb + ksel;
        FragU a = load_frag(arow, lk);
        FragU b[4];
#pragma unroll
        for (int g = 0; g < 4; ++g) b[g] = load_frag(brow[g], lk);
#pragma unroll
        for (int g = 0; g < 4; ++g) acc[g] = wmma_bf16(a, b[g], acc[g]);
    }

    int hh = ncol;
#pragma unroll
    for (int r = 0; r < 8; ++r) {
        int m = m0 + r + ((lane >> 4) << 3);
        const float* xrow = xw + ((size_t)m * T + t) * LG;
        float pi = acc[0][r] + xrow[hh];
        float pf = acc[1][r] + xrow[H + hh];
        float po = acc[2][r] + xrow[2 * H + hh];
        float pg = acc[3][r] + xrow[3 * H + hh];
        float iv = sigmoidf_(pi);
        float fv = sigmoidf_(pf);
        float ov = sigmoidf_(po);
        float gv = tanhf(pg);
        size_t ci = (size_t)m * H + hh;
        float cn = fv * cbuf[ci] + iv * gv;
        float hn = ov * tanhf(cn);
        cbuf[ci] = cn;
        out[((size_t)m * T + t) * H + hh] = hn;
        hout[ci] = f2bf(hn);
    }
}

// ---------------------------------------------------------------------------
extern "C" void kernel_launch(void* const* d_in, const int* in_sizes, int n_in,
                              void* d_out, int out_size, void* d_ws, size_t ws_size,
                              hipStream_t stream) {
    (void)in_sizes; (void)n_in; (void)out_size; (void)ws_size;
    const float* x  = (const float*)d_in[0];  // N*T*D
    const float* h0 = (const float*)d_in[1];  // N*H
    const float* Wx = (const float*)d_in[2];  // D*4H
    const float* Wh = (const float*)d_in[3];  // H*4H
    const float* b  = (const float*)d_in[4];  // 4H
    float* out = (float*)d_out;               // N*T*H

    char* ws = (char*)d_ws;
    size_t off = 0;
    auto walloc = [&](size_t bytes) -> void* {
        void* p = ws + off;
        off = (off + bytes + 255) & ~(size_t)255;
        return p;
    };
    float*          xw   = (float*)walloc((size_t)LN * LT * LG * sizeof(float));          // 512 MB
    unsigned short* xbf  = (unsigned short*)walloc((size_t)LN * LT * LD * sizeof(short)); //  64 MB
    unsigned short* wxT  = (unsigned short*)walloc((size_t)LG * LD * sizeof(short));      //   8 MB
    unsigned short* whT  = (unsigned short*)walloc((size_t)LG * LH * sizeof(short));      //   8 MB
    unsigned short* hA   = (unsigned short*)walloc((size_t)LN * LH * sizeof(short));
    unsigned short* hB   = (unsigned short*)walloc((size_t)LN * LH * sizeof(short));
    float*          cbuf = (float*)walloc((size_t)LN * LH * sizeof(float));

    hipMemsetAsync(cbuf, 0, (size_t)LN * LH * sizeof(float), stream);

    // Stage bf16 operands.
    {
        int n = LN * LT * LD;
        cvt_f32_bf16_kernel<<<(n + 255) / 256, 256, 0, stream>>>(x, xbf, n);
    }
    {
        int n = LN * LH;
        cvt_f32_bf16_kernel<<<(n + 255) / 256, 256, 0, stream>>>(h0, hA, n);
    }
    transpose_cvt_kernel<<<dim3(LG / 32, LD / 32), dim3(32, 8), 0, stream>>>(Wx, wxT, LD, LG);
    transpose_cvt_kernel<<<dim3(LG / 32, LH / 32), dim3(32, 8), 0, stream>>>(Wh, whT, LH, LG);

    // Parallel input GEMM: xW = x @ Wx + b.
    // waves = (M/16)*(G/64) = 2048*64 = 131072 -> 16384 blocks of 8 waves.
    gemm_xw_kernel<<<16384, 256, 0, stream>>>(xbf, wxT, b, xw);

    // Sequential recurrence, one fused kernel per timestep (ping-pong h).
    for (int t = 0; t < LT; ++t) {
        const unsigned short* hin = (t & 1) ? hB : hA;
        unsigned short*      hout = (t & 1) ? hA : hB;
        lstm_step_kernel<<<32, 256, 0, stream>>>(hin, whT, xw, cbuf, out, hout, t);
    }
}
